// EdgeEmbedding_87900800680239
// MI455X (gfx1250) — matrix operand order repaired and checked by reading, not compiled, not run on hardware
//
#include <hip/hip_runtime.h>

typedef __attribute__((ext_vector_type(16))) _Float16 v16h;
typedef __attribute__((ext_vector_type(8)))  _Float16 v8h;
typedef __attribute__((ext_vector_type(8)))  float    v8f;

// Problem constants (fixed by the reference)
#define D_ATOM   256
#define D_OUT    256
#define K_TOTAL  768          // 2*D_ATOM + D_EDGE
#define M_TILE   128          // edges per workgroup
#define K_CHUNK  32           // K per WMMA f16 instruction
#define N_CHUNKS (K_TOTAL / K_CHUNK)   // 24
#define THREADS  512          // 16 waves (wave32)
#define A_STRIDE 40           // halves per LDS row (32 data + 8 pad) = 80 bytes
#define B_STRIDE 40
#define A_SZ     (M_TILE * A_STRIDE)   // halves per A buffer
#define B_SZ     (D_OUT  * B_STRIDE)   // halves per B buffer

__global__ __launch_bounds__(THREADS) void edge_embed_wmma(
    const float* __restrict__ hmat, const float* __restrict__ m_rbf,
    const int* __restrict__ idnb_a, const int* __restrict__ idnb_c,
    const float* __restrict__ W, float* __restrict__ out)
{
  // double-buffered staging: 2 * (10 KB + 20 KB) = 60 KB LDS
  __shared__ __align__(16) _Float16 As[2 * A_SZ]; // row-major [row][k]
  __shared__ __align__(16) _Float16 Bt[2 * B_SZ]; // N-major  [n][k]

  const int t    = threadIdx.x;
  const int lane = t & 31;
  const int wave = t >> 5;
  const int wm   = wave & 3;    // 4 M-subtiles of 32 rows
  const int wn   = wave >> 2;   // 4 N-subtiles of 64 cols

  const int E0 = blockIdx.x * M_TILE;

  // ---- A-staging role: 4 threads per edge row, each covers 8 floats of the chunk
  const int rowA = t >> 2;            // 0..127
  const int kq   = (t & 3) * 8;       // 0,8,16,24 within the 32-wide chunk
  const int e    = E0 + rowA;
  const float* pa = hmat  + (size_t)idnb_a[e] * D_ATOM;
  const float* pc = hmat  + (size_t)idnb_c[e] * D_ATOM;
  const float* pm = m_rbf + (size_t)e        * D_ATOM;

  // ---- B-staging role: 2 threads per output column, each covers 16 K values
  const int nB   = t & 255;           // output column 0..255
  const int ksub = t >> 8;            // 0..1 -> K range ksub*16 .. +15

  const int hsel = lane >> 4;         // which half-wave
  const int l16  = lane & 15;

  // load + convert one K-chunk worth of A/B data into registers
  auto load_chunk = [&](int kc, v8h& ah, v8h& bh0, v8h& bh1) {
    const int k0 = kc * K_CHUNK;
    const float* srcA = (k0 < D_ATOM)     ? (pa + k0)
                      : (k0 < 2 * D_ATOM) ? (pc + (k0 - D_ATOM))
                                          : (pm + (k0 - 2 * D_ATOM));
    const float4 f0 = *(const float4*)(srcA + kq);
    const float4 f1 = *(const float4*)(srcA + kq + 4);
    ah[0] = (_Float16)f0.x; ah[1] = (_Float16)f0.y;
    ah[2] = (_Float16)f0.z; ah[3] = (_Float16)f0.w;
    ah[4] = (_Float16)f1.x; ah[5] = (_Float16)f1.y;
    ah[6] = (_Float16)f1.z; ah[7] = (_Float16)f1.w;

    const float* wp = W + (size_t)(k0 + ksub * 16) * D_OUT + nB; // coalesced 4B/lane
    float bv[16];
    #pragma unroll
    for (int i = 0; i < 16; ++i) bv[i] = wp[(size_t)i * D_OUT];
    #pragma unroll
    for (int i = 0; i < 8; ++i) {
      bh0[i] = (_Float16)bv[i];
      bh1[i] = (_Float16)bv[8 + i];
    }
  };

  v8f acc[2][4];
  #pragma unroll
  for (int i = 0; i < 2; ++i)
    #pragma unroll
    for (int j = 0; j < 4; ++j)
      #pragma unroll
      for (int v = 0; v < 8; ++v) acc[i][j][v] = 0.0f;

  v8h ah, bh0, bh1;
  load_chunk(0, ah, bh0, bh1);

  #pragma unroll
  for (int kc = 0; kc < N_CHUNKS; ++kc) {
    // publish chunk kc into buffer kc&1 (safe: all waves last read this buffer
    // before the barrier of iteration kc-1)
    _Float16* as = As + (kc & 1) * A_SZ;
    _Float16* bt = Bt + (kc & 1) * B_SZ;
    *(v8h*)(&as[rowA * A_STRIDE + kq])            = ah;   // ds_store_b128
    *(v8h*)(&bt[nB   * B_STRIDE + ksub * 16])     = bh0;  // transposed: [n][k]
    *(v8h*)(&bt[nB   * B_STRIDE + ksub * 16 + 8]) = bh1;

    // prefetch chunk kc+1 into registers (overlaps the WMMAs below)
    v8h nah = ah, nb0 = bh0, nb1 = bh1;
    if (kc + 1 < N_CHUNKS) load_chunk(kc + 1, nah, nb0, nb1);

    __syncthreads();   // stores of chunk kc visible; prior reads of this buffer done

    // ---- A fragments: lane holds row (l16); halves {0-7,16-23} or {8-15,24-31}
    v16h afrag[2];
    #pragma unroll
    for (int fa = 0; fa < 2; ++fa) {
      const int row = wm * 32 + fa * 16 + l16;
      v8h lo = *(const v8h*)(&as[row * A_STRIDE + hsel * 8]);
      v8h hi = *(const v8h*)(&as[row * A_STRIDE + hsel * 8 + 16]);
      afrag[fa] = __builtin_shufflevector(lo, hi,
          0, 1, 2, 3, 4, 5, 6, 7, 8, 9, 10, 11, 12, 13, 14, 15);
    }

    // ---- B fragments: lane holds column (l16); K {0-7,16-23} (lanes<16) / {8-15,24-31} (lanes>=16)
    #pragma unroll
    for (int fb = 0; fb < 4; ++fb) {
      const int col = wn * 64 + fb * 16 + l16;
      v8h lo = *(const v8h*)(&bt[col * B_STRIDE + hsel * 8]);
      v8h hi = *(const v8h*)(&bt[col * B_STRIDE + hsel * 8 + 16]);
      v16h bfrag = __builtin_shufflevector(lo, hi,
          0, 1, 2, 3, 4, 5, 6, 7, 8, 9, 10, 11, 12, 13, 14, 15);
      #pragma unroll
      for (int fa = 0; fa < 2; ++fa) {
        acc[fa][fb] = __builtin_amdgcn_wmma_f32_16x16x32_f16(
            /*neg_a=*/false, afrag[fa], /*neg_b=*/false, bfrag,
            /*c_mod=*/(short)0, acc[fa][fb],
            /*reuse_a=*/false, /*reuse_b=*/false);
      }
    }

    ah = nah; bh0 = nb0; bh1 = nb1;
  }

  // ---- epilogue: SiLU + store (C/D layout: lane=column, vgpr v = row within half)
  #pragma unroll
  for (int fa = 0; fa < 2; ++fa) {
    #pragma unroll
    for (int fb = 0; fb < 4; ++fb) {
      const int col = wn * 64 + fb * 16 + l16;
      #pragma unroll
      for (int v = 0; v < 8; ++v) {
        const int row = E0 + wm * 32 + fa * 16 + hsel * 8 + v;
        const float x = acc[fa][fb][v];
        const float y = x / (1.0f + __expf(-x));   // silu
        out[(size_t)row * D_OUT + col] = y;
      }
    }
  }
}

extern "C" void kernel_launch(void* const* d_in, const int* in_sizes, int n_in,
                              void* d_out, int out_size, void* d_ws, size_t ws_size,
                              hipStream_t stream) {
  const float* hmat  = (const float*)d_in[0];
  const float* m_rbf = (const float*)d_in[1];
  const int*   ia    = (const int*)d_in[2];
  const int*   ic    = (const int*)d_in[3];
  const float* W     = (const float*)d_in[4];
  float* out = (float*)d_out;

  const int nEdges = in_sizes[2];          // 320000, multiple of M_TILE
  const int grid   = nEdges / M_TILE;      // 2500 workgroups

  edge_embed_wmma<<<grid, THREADS, 0, stream>>>(hmat, m_rbf, ia, ic, W, out);
}
